// MultiHeadAttention_36069135352205
// MI455X (gfx1250) — compile-verified, matrix operand order checked by reading, not run on hardware
//
#include <hip/hip_runtime.h>
#include <hip/hip_bf16.h>

// ---------------------------------------------------------------------------
// MHA forward for MI455X (gfx1250), wave32, bf16 WMMA with f32 accumulate.
// Dims: B=2, L=2048, D_MODEL=1024, H=16, DK=64.
// ---------------------------------------------------------------------------

typedef unsigned short ushort_t;
typedef __attribute__((ext_vector_type(16))) __bf16 v16bf;
typedef __attribute__((ext_vector_type(8)))  float  v8f;

#define DMODEL 1024
#define NHEAD  16
#define DK     64
#define BATCH  2
#define SEQL   2048
#define MROWS  (BATCH * SEQL)   // 4096

// f32 -> bf16 with round-to-nearest-even (bit trick), as raw u16
__device__ __forceinline__ ushort_t f2bf_u(float f) {
  unsigned u = __float_as_uint(f);
  u += 0x7FFFu + ((u >> 16) & 1u);
  return (ushort_t)(u >> 16);
}

// load 16 contiguous bf16 (32B) as an operand fragment slice
__device__ __forceinline__ v16bf load_frag(const ushort_t* p) {
  v16bf f;
  __builtin_memcpy(&f, p, 32);
  return f;
}

__device__ __forceinline__ v8f wmma_bf16(v16bf a, v16bf b, v8f c) {
  // D = A(16x32) * B(32x16) + C(16x16), f32 accum
  return __builtin_amdgcn_wmma_f32_16x16x32_bf16(
      /*neg_a=*/false, a, /*neg_b=*/false, b,
      /*c_mod=*/(short)0, c, /*reuse_a=*/false, /*reuse_b=*/false);
}

// ---------------------------------------------------------------------------
// f32 -> bf16 conversion kernel
// ---------------------------------------------------------------------------
__global__ void cvt_f32_bf16(const float* __restrict__ in,
                             ushort_t* __restrict__ out, int n) {
  int i = blockIdx.x * blockDim.x + threadIdx.x;
  if (i < n) out[i] = f2bf_u(in[i]);
}

// ---------------------------------------------------------------------------
// GEMM: Y = X(bf16, MxK) * W(bf16, NxK)^T + bias
// MODE 0: f32 row-major output (ld = N)
// MODE 1: bf16 output in (B, H, L, DK) head layout   (Q, K)
// MODE 2: bf16 output in (B, H, DK, L) head-transposed layout (V)
// Block: 256 threads = 8 waves in 2(M) x 4(N); wave tile 64x32; WG tile 128x128.
// ---------------------------------------------------------------------------
template <int MODE>
__global__ __launch_bounds__(256)
void gemm_xwt(const ushort_t* __restrict__ A, const ushort_t* __restrict__ W,
              const float* __restrict__ bias, void* __restrict__ outp,
              int Mn, int Nn, int Kn) {
  const int lane = threadIdx.x & 31;
  const int wave = threadIdx.x >> 5;
  const int bm = blockIdx.x * 128, bn = blockIdx.y * 128;
  const int wm = (wave >> 2) * 64, wn = (wave & 3) * 32;
  const int lr = lane & 15;
  const int kh = (lane >> 4) * 16;   // K-half select per lane group

  int ra[4], rb[2];
#pragma unroll
  for (int i = 0; i < 4; ++i) ra[i] = bm + wm + i * 16 + lr;
#pragma unroll
  for (int j = 0; j < 2; ++j) rb[j] = bn + wn + j * 16 + lr;

  const v8f zero = {0.f, 0.f, 0.f, 0.f, 0.f, 0.f, 0.f, 0.f};
  v8f acc[4][2];
#pragma unroll
  for (int i = 0; i < 4; ++i)
#pragma unroll
    for (int j = 0; j < 2; ++j) acc[i][j] = zero;

  for (int k0 = 0; k0 < Kn; k0 += 32) {
    v16bf af[4], bfr[2];
#pragma unroll
    for (int i = 0; i < 4; ++i)
      af[i] = load_frag(A + (size_t)ra[i] * Kn + k0 + kh);
#pragma unroll
    for (int j = 0; j < 2; ++j)
      bfr[j] = load_frag(W + (size_t)rb[j] * Kn + k0 + kh);

    if (k0 + 32 < Kn) {  // prefetch next K-tile -> global_prefetch_b8
      __builtin_prefetch(A + (size_t)ra[0] * Kn + (k0 + 32) + kh, 0, 0);
      __builtin_prefetch(W + (size_t)rb[0] * Kn + (k0 + 32) + kh, 0, 0);
    }

#pragma unroll
    for (int i = 0; i < 4; ++i)
#pragma unroll
      for (int j = 0; j < 2; ++j)
        acc[i][j] = wmma_bf16(af[i], bfr[j], acc[i][j]);
  }

  // Store (C/D layout: N = lane&15, M = vgpr + 8*(lane>>4))
#pragma unroll
  for (int j = 0; j < 2; ++j) {
    const int n = bn + wn + j * 16 + lr;
    const float bv = bias ? bias[n] : 0.0f;
#pragma unroll
    for (int i = 0; i < 4; ++i) {
#pragma unroll
      for (int r = 0; r < 8; ++r) {
        const int m = bm + wm + i * 16 + r + (lane >> 4) * 8;
        const float v = acc[i][j][r] + bv;
        if (MODE == 0) {
          ((float*)outp)[(size_t)m * Nn + n] = v;
        } else {
          const int b = m >> 11, l = m & (SEQL - 1);
          const int h = n >> 6, d = n & (DK - 1);
          const size_t idx =
              (MODE == 1)
                  ? (((size_t)(b * NHEAD + h) * SEQL + l) * DK + d)
                  : (((size_t)(b * NHEAD + h) * DK + d) * SEQL + l);
          ((ushort_t*)outp)[idx] = f2bf_u(v);
        }
      }
    }
  }
}

// ---------------------------------------------------------------------------
// Scores: S[b,h,q,k] = (Q . K) / 8, masked; write f32 into attn buffer.
// Block: 256 threads = 8 waves in 4(M) x 2(N); wave tile 32x64; WG tile 128x128.
// Grid: (L/128, L/128, B*H)
// ---------------------------------------------------------------------------
__global__ __launch_bounds__(256)
void scores_kernel(const ushort_t* __restrict__ Qh,
                   const ushort_t* __restrict__ Kh,
                   const int* __restrict__ mask,
                   float* __restrict__ attn) {
  const int lane = threadIdx.x & 31;
  const int wave = threadIdx.x >> 5;
  const int bh = blockIdx.z;
  const int bidx = bh >> 4;
  const int bm = blockIdx.x * 128, bn = blockIdx.y * 128;
  const int wm = (wave >> 1) * 32, wn = (wave & 1) * 64;
  const int lr = lane & 15;
  const int kh = (lane >> 4) * 16;

  const ushort_t* Q = Qh + (size_t)bh * SEQL * DK;
  const ushort_t* Kp = Kh + (size_t)bh * SEQL * DK;

  const v8f zero = {0.f, 0.f, 0.f, 0.f, 0.f, 0.f, 0.f, 0.f};
  v8f acc[2][4];
#pragma unroll
  for (int i = 0; i < 2; ++i)
#pragma unroll
    for (int j = 0; j < 4; ++j) acc[i][j] = zero;

#pragma unroll
  for (int k0 = 0; k0 < DK; k0 += 32) {
    v16bf af[2], bfr[4];
#pragma unroll
    for (int i = 0; i < 2; ++i)
      af[i] = load_frag(Q + (size_t)(bm + wm + i * 16 + lr) * DK + k0 + kh);
#pragma unroll
    for (int j = 0; j < 4; ++j)
      bfr[j] = load_frag(Kp + (size_t)(bn + wn + j * 16 + lr) * DK + k0 + kh);
#pragma unroll
    for (int i = 0; i < 2; ++i)
#pragma unroll
      for (int j = 0; j < 4; ++j)
        acc[i][j] = wmma_bf16(af[i], bfr[j], acc[i][j]);
  }

  float* Sp = attn + (size_t)bh * SEQL * SEQL;
#pragma unroll
  for (int j = 0; j < 4; ++j) {
    const int n = bn + wn + j * 16 + lr;
    const int mk = mask[bidx * SEQL + n];
#pragma unroll
    for (int i = 0; i < 2; ++i) {
#pragma unroll
      for (int r = 0; r < 8; ++r) {
        const int m = bm + wm + i * 16 + r + (lane >> 4) * 8;
        const float v = mk ? acc[i][j][r] * 0.125f : -1.0e9f;
        Sp[(size_t)m * SEQL + n] = v;
      }
    }
  }
}

// ---------------------------------------------------------------------------
// Row softmax over K (2048 cols), in place. One WG (256 threads) per row.
// ---------------------------------------------------------------------------
__global__ __launch_bounds__(256)
void softmax_kernel(float* __restrict__ attn) {
  float* p = attn + (size_t)blockIdx.x * SEQL;
  const int t = threadIdx.x;

  float v[8];
  float mx = -3.4e38f;
#pragma unroll
  for (int i = 0; i < 8; ++i) {
    v[i] = p[t + i * 256];
    mx = fmaxf(mx, v[i]);
  }

  __shared__ float red[256];
  red[t] = mx;
  __syncthreads();
  for (int s = 128; s > 0; s >>= 1) {
    if (t < s) red[t] = fmaxf(red[t], red[t + s]);
    __syncthreads();
  }
  mx = red[0];
  __syncthreads();

  float sum = 0.f;
#pragma unroll
  for (int i = 0; i < 8; ++i) {
    v[i] = __expf(v[i] - mx);
    sum += v[i];
  }
  red[t] = sum;
  __syncthreads();
  for (int s = 128; s > 0; s >>= 1) {
    if (t < s) red[t] += red[t + s];
    __syncthreads();
  }
  const float inv = 1.0f / red[0];
#pragma unroll
  for (int i = 0; i < 8; ++i) p[t + i * 256] = v[i] * inv;
}

// ---------------------------------------------------------------------------
// attn @ V : per (b,h), out(2048 x 64) = attn(2048 x 2048, f32->bf16) * V.
// V is stored transposed per head (DK x L), so B-operand rows are contiguous.
// Block: 128 threads = 4 waves stacked in M; wave tile 32x64; WG tile 128x64.
// Output ctx in (B, L, D_MODEL) bf16 row-major.
// ---------------------------------------------------------------------------
__global__ __launch_bounds__(128)
void av_kernel(const float* __restrict__ attn,
               const ushort_t* __restrict__ Vt,
               ushort_t* __restrict__ ctx) {
  const int lane = threadIdx.x & 31;
  const int wave = threadIdx.x >> 5;
  const int bh = blockIdx.z;
  const int b = bh >> 4, h = bh & 15;
  const int bm = blockIdx.x * 128;
  const int wm = wave * 32;
  const int lr = lane & 15;
  const int kh = (lane >> 4) * 16;

  const float* Ap = attn + (size_t)bh * SEQL * SEQL;
  const ushort_t* Vp = Vt + (size_t)bh * DK * SEQL;

  const v8f zero = {0.f, 0.f, 0.f, 0.f, 0.f, 0.f, 0.f, 0.f};
  v8f acc[2][4];
#pragma unroll
  for (int i = 0; i < 2; ++i)
#pragma unroll
    for (int j = 0; j < 4; ++j) acc[i][j] = zero;

  const int qrow[2] = {bm + wm + lr, bm + wm + 16 + lr};
  int vrow[4];
#pragma unroll
  for (int j = 0; j < 4; ++j) vrow[j] = j * 16 + lr;

  for (int k0 = 0; k0 < SEQL; k0 += 32) {
    v16bf af[2];
#pragma unroll
    for (int i = 0; i < 2; ++i) {
      const float* p = Ap + (size_t)qrow[i] * SEQL + k0 + kh;
      const float4 f0 = ((const float4*)p)[0];
      const float4 f1 = ((const float4*)p)[1];
      const float4 f2 = ((const float4*)p)[2];
      const float4 f3 = ((const float4*)p)[3];
      const float fv[16] = {f0.x, f0.y, f0.z, f0.w, f1.x, f1.y, f1.z, f1.w,
                            f2.x, f2.y, f2.z, f2.w, f3.x, f3.y, f3.z, f3.w};
      ushort_t tb[16];
#pragma unroll
      for (int e = 0; e < 16; ++e) tb[e] = f2bf_u(fv[e]);
      __builtin_memcpy(&af[i], tb, 32);
    }

    v16bf bfr[4];
#pragma unroll
    for (int j = 0; j < 4; ++j)
      bfr[j] = load_frag(Vp + (size_t)vrow[j] * SEQL + k0 + kh);

    if (k0 + 32 < SEQL)
      __builtin_prefetch(Ap + (size_t)qrow[0] * SEQL + (k0 + 32) + kh, 0, 0);

#pragma unroll
    for (int i = 0; i < 2; ++i)
#pragma unroll
      for (int j = 0; j < 4; ++j)
        acc[i][j] = wmma_bf16(af[i], bfr[j], acc[i][j]);
  }

#pragma unroll
  for (int j = 0; j < 4; ++j) {
    const int n = j * 16 + lr;
#pragma unroll
    for (int i = 0; i < 2; ++i) {
#pragma unroll
      for (int r = 0; r < 8; ++r) {
        const int m = bm + wm + i * 16 + r + (lane >> 4) * 8;
        ctx[(size_t)(b * SEQL + m) * DMODEL + h * DK + n] =
            f2bf_u(acc[i][j][r]);
      }
    }
  }
}

// ---------------------------------------------------------------------------
// Host-side launcher
// ---------------------------------------------------------------------------
extern "C" void kernel_launch(void* const* d_in, const int* in_sizes, int n_in,
                              void* d_out, int out_size, void* d_ws,
                              size_t ws_size, hipStream_t stream) {
  (void)in_sizes; (void)n_in; (void)out_size; (void)ws_size;

  const float* x    = (const float*)d_in[0];
  const int*   mask = (const int*)d_in[1];
  const float* Wq   = (const float*)d_in[2];
  const float* bq   = (const float*)d_in[3];
  const float* Wk   = (const float*)d_in[4];
  const float* bk   = (const float*)d_in[5];
  const float* Wv   = (const float*)d_in[6];
  const float* bv   = (const float*)d_in[7];
  const float* Wo   = (const float*)d_in[8];
  const float* bo   = (const float*)d_in[9];

  const size_t SZ_X = (size_t)MROWS * DMODEL;   // 4M elems
  const size_t SZ_W = (size_t)DMODEL * DMODEL;  // 1M elems

  ushort_t* ws  = (ushort_t*)d_ws;
  ushort_t* xb  = ws;               // x in bf16            (4M)
  ushort_t* Wqb = xb + SZ_X;        // weights in bf16      (1M each)
  ushort_t* Wkb = Wqb + SZ_W;
  ushort_t* Wvb = Wkb + SZ_W;
  ushort_t* Wob = Wvb + SZ_W;
  ushort_t* Qh  = Wob + SZ_W;       // (B,H,L,DK)           (4M)
  ushort_t* Kh  = Qh + SZ_X;        // (B,H,L,DK)           (4M)
  ushort_t* Vt  = Kh + SZ_X;        // (B,H,DK,L)           (4M)
  ushort_t* ctx = Vt + SZ_X;        // (B,L,D)              (4M)

  float* out  = (float*)d_out;           // (B,L,D) f32
  float* attn = out + SZ_X;              // (B,H,L,L) f32

  // 1) casts to bf16
  cvt_f32_bf16<<<(int)(SZ_X / 256), 256, 0, stream>>>(x, xb, (int)SZ_X);
  cvt_f32_bf16<<<(int)(SZ_W / 256), 256, 0, stream>>>(Wq, Wqb, (int)SZ_W);
  cvt_f32_bf16<<<(int)(SZ_W / 256), 256, 0, stream>>>(Wk, Wkb, (int)SZ_W);
  cvt_f32_bf16<<<(int)(SZ_W / 256), 256, 0, stream>>>(Wv, Wvb, (int)SZ_W);
  cvt_f32_bf16<<<(int)(SZ_W / 256), 256, 0, stream>>>(Wo, Wob, (int)SZ_W);

  // 2) Q/K/V projections (WMMA bf16). Q,K -> head layout; V -> head-transposed.
  dim3 gproj(MROWS / 128, DMODEL / 128, 1);
  gemm_xwt<1><<<gproj, 256, 0, stream>>>(xb, Wqb, bq, Qh, MROWS, DMODEL, DMODEL);
  gemm_xwt<1><<<gproj, 256, 0, stream>>>(xb, Wkb, bk, Kh, MROWS, DMODEL, DMODEL);
  gemm_xwt<2><<<gproj, 256, 0, stream>>>(xb, Wvb, bv, Vt, MROWS, DMODEL, DMODEL);

  // 3) scores (scale + mask fused), f32 into d_out attn region
  scores_kernel<<<dim3(SEQL / 128, SEQL / 128, BATCH * NHEAD), 256, 0, stream>>>(
      Qh, Kh, mask, attn);

  // 4) row softmax in place
  softmax_kernel<<<dim3(BATCH * NHEAD * SEQL), 256, 0, stream>>>(attn);

  // 5) attn @ V -> ctx (bf16, (B,L,D))
  av_kernel<<<dim3(SEQL / 128, 1, BATCH * NHEAD), 128, 0, stream>>>(attn, Vt, ctx);

  // 6) output projection -> f32 out
  gemm_xwt<0><<<gproj, 256, 0, stream>>>(ctx, Wob, bo, out, MROWS, DMODEL, DMODEL);
}
